// MultiHeadAttention_5128190951704
// MI455X (gfx1250) — matrix operand order repaired and checked by reading, not compiled, not run on hardware
//
#include <hip/hip_runtime.h>
#include <hip/hip_bf16.h>
#include <math.h>

// ---------------------------------------------------------------------------
// Problem constants (from reference): B=16, C=512, H=W=64 -> HW=4096, nh=8,
// hd=64, 3C=1536.  All fp32 in / fp32 out; GEMMs internally in bf16 WMMA.
// ---------------------------------------------------------------------------
#define BATCH 16
#define CCH   512
#define HW    4096
#define NH    8
#define HD    64
#define M_QKV 1536

typedef __attribute__((ext_vector_type(16))) __bf16 v16bf;
typedef __attribute__((ext_vector_type(8)))  __bf16 v8bf;
typedef __attribute__((ext_vector_type(8)))  float  v8f;

static __device__ __forceinline__ v8f wmma_bf16(v16bf a, v16bf b, v8f c) {
  // (neg_a, A, neg_b, B, c_mod, C, reuse_a, reuse_b)
  return __builtin_amdgcn_wmma_f32_16x16x32_bf16(false, a, false, b,
                                                 (short)0, c, false, false);
}
static __device__ __forceinline__ v16bf cat8(v8bf lo, v8bf hi) {
  return __builtin_shufflevector(lo, hi, 0,1,2,3,4,5,6,7,8,9,10,11,12,13,14,15);
}

// ---------------------------------------------------------------------------
// Tensor Data Mover support (gfx1250). Guarded so either toolchain compiles.
// ---------------------------------------------------------------------------
#if defined(__has_builtin)
#  if __has_builtin(__builtin_amdgcn_tensor_load_to_lds) && \
      __has_builtin(__builtin_amdgcn_s_wait_tensorcnt)
#    define HAVE_TDM 1
#  endif
#endif
#ifndef HAVE_TDM
#  define HAVE_TDM 0
#endif

#if HAVE_TDM
typedef __attribute__((ext_vector_type(4))) unsigned int u32x4;
typedef __attribute__((ext_vector_type(4))) int          i32x4;
typedef __attribute__((ext_vector_type(8))) int          i32x8;

// Issue one TDM 2D tile load: 128 rows x 32 bf16 (64B/row), row stride =
// row_stride elements, into contiguous LDS at lds_dst.  D# per ISA 08 §8:
//   group0: count=1 | lds_addr | global_addr[56:0] | type=2
//   group1: data_size=1(2B), tensor_dim0=32, tensor_dim1=128,
//           tile_dim0=32, tile_dim1=128, tensor_dim0_stride=row_stride
static __device__ __forceinline__ void tdm_load_a_tile(const __bf16* gsrc,
                                                       void* lds_dst,
                                                       int row_stride) {
  unsigned long long ga = (unsigned long long)(size_t)gsrc;
  unsigned lds_off = (unsigned)(size_t)lds_dst;  // flat LDS addr low 32 bits
  u32x4 g0;
  g0[0] = 1u;                                           // count=1, user mode
  g0[1] = lds_off;                                      // lds_addr (bytes)
  g0[2] = (unsigned)ga;                                 // global_addr[31:0]
  g0[3] = ((unsigned)(ga >> 32) & 0x01ffffffu) | (2u << 30);  // [56:32],type=2
  i32x8 g1;
  g1[0] = (int)(1u << 16);      // workgroup_mask=0, data_size=1 (2 bytes)
  g1[1] = (int)(32u << 16);     // tensor_dim0 = 32 (bits[63:48])
  g1[2] = (int)(128u << 16);    // tensor_dim1 = 128 (bits[95:80])
  g1[3] = (int)(32u << 16);     // tile_dim0 = 32 (bits[127:112])
  g1[4] = 128;                  // tile_dim1 = 128 (bits[143:128])
  g1[5] = row_stride;           // tensor_dim0_stride[31:0] (elements)
  g1[6] = 0;
  g1[7] = 0;
  i32x4 gz = (i32x4){0, 0, 0, 0};
#if __clang_major__ >= 23
  i32x8 gz8 = (i32x8){0, 0, 0, 0, 0, 0, 0, 0};
  __builtin_amdgcn_tensor_load_to_lds(g0, g1, gz, gz, gz8, 0);
#else
  __builtin_amdgcn_tensor_load_to_lds(g0, g1, gz, gz, 0);
#endif
}
#endif  // HAVE_TDM

// ---------------------------------------------------------------------------
// fp32 -> bf16 conversion (grid-stride)
// ---------------------------------------------------------------------------
__global__ void cvt_bf16_kernel(const float* __restrict__ in,
                                __bf16* __restrict__ out, int n) {
  for (int i = blockIdx.x * blockDim.x + threadIdx.x; i < n;
       i += gridDim.x * blockDim.x) {
    out[i] = (__bf16)in[i];
  }
}

// ---------------------------------------------------------------------------
// Batched GEMM: C[z] = A[MxK] * B[z][KxN]  (A shared across batch).
// Block tile 128x128, 256 threads = 8 waves, each wave 64(M) x 32(N):
// 4x2 fragments of v_wmma_f32_16x16x32_bf16, K-step 32, double-buffered LDS.
// A tile is DMA'd by the Tensor Data Mover (wave 0 issues, s_wait_tensorcnt
// before the barrier); B tile is staged column-major [n][k] so the B fragment
// (column n, 16 consecutive K) is one contiguous 32B LDS read.
// ---------------------------------------------------------------------------
template <bool BF16OUT>
__global__ __launch_bounds__(256)
void gemm_bf16_kernel(const __bf16* __restrict__ A,
                      const __bf16* __restrict__ B,
                      void* __restrict__ Cout,
                      const float* __restrict__ bias,
                      int M, int N, int K) {
  __shared__ __align__(32) __bf16 As[2][128 * 32];
  __shared__ __align__(32) __bf16 Bs[2][128 * 32];  // [n][k] column-major

  const int tid  = threadIdx.x;
  const int lane = tid & 31;
  const int wid  = tid >> 5;
  const int lh   = lane >> 4;   // K-half selector (lanes 16-31)
  const int l16  = lane & 15;

  const int nBase = blockIdx.x * 128;
  const int mBase = blockIdx.y * 128;
  const int z     = blockIdx.z;
  const __bf16* Bz = B + (size_t)z * (size_t)K * (size_t)N;

  const int mOff = (wid >> 2) * 64;  // 2 wave rows
  const int nOff = (wid & 3) * 32;   // 4 wave cols

  v8f acc[4][2];
#pragma unroll
  for (int mm = 0; mm < 4; ++mm)
#pragma unroll
    for (int nn = 0; nn < 2; ++nn) acc[mm][nn] = (v8f){0,0,0,0,0,0,0,0};

  const int ar = tid >> 1, ach = (tid & 1) * 16;   // A-tile fallback coords
  const int bk = tid >> 3, bc0 = (tid & 7) * 16;   // B-tile load coords

  // ---- prologue: stage tile 0 into buffer 0 ----
#if HAVE_TDM
  if (wid == 0) tdm_load_a_tile(A + (size_t)mBase * K, &As[0][0], K);
#else
  {
    v16bf av = *(const v16bf*)(A + (size_t)(mBase + ar) * K + ach);
    *(v16bf*)(&As[0][ar * 32 + ach]) = av;
  }
#endif
  {
    v16bf bv = *(const v16bf*)(Bz + (size_t)bk * N + nBase + bc0);
#pragma unroll
    for (int j = 0; j < 16; ++j) Bs[0][(bc0 + j) * 32 + bk] = bv[j];
  }
#if HAVE_TDM
  if (wid == 0) __builtin_amdgcn_s_wait_tensorcnt(0);
#endif
  __syncthreads();

  for (int kk = 0; kk < K; kk += 32) {
    const int cur = (kk >> 5) & 1;
    const int nxt = cur ^ 1;
    const bool more = (kk + 32) < K;

    if (more) {
      // stage next K-tile while current one is consumed
#if HAVE_TDM
      if (wid == 0)
        tdm_load_a_tile(A + (size_t)mBase * K + kk + 32, &As[nxt][0], K);
#else
      v16bf av = *(const v16bf*)(A + (size_t)(mBase + ar) * K + kk + 32 + ach);
      *(v16bf*)(&As[nxt][ar * 32 + ach]) = av;
#endif
      v16bf bv = *(const v16bf*)(Bz + (size_t)(kk + 32 + bk) * N + nBase + bc0);
#pragma unroll
      for (int j = 0; j < 16; ++j) Bs[nxt][(bc0 + j) * 32 + bk] = bv[j];
      if (kk + 64 < K)  // hint tile-after-next (global_prefetch_b8)
        __builtin_prefetch(Bz + (size_t)(kk + 64 + bk) * N + nBase + bc0, 0, 1);
    }

    // ---- compute on current buffer ----
    v16bf afrag[4];
#pragma unroll
    for (int mm = 0; mm < 4; ++mm) {
      const __bf16* ap = &As[cur][(mOff + mm * 16 + l16) * 32 + 16 * lh];
      v8bf lo = *(const v8bf*)ap;          // K = 16*lh .. 16*lh+7
      v8bf hi = *(const v8bf*)(ap + 8);    // K = 16*lh+8 .. 16*lh+15
      afrag[mm] = cat8(lo, hi);
    }
    v16bf bfrag[2];
#pragma unroll
    for (int nn = 0; nn < 2; ++nn)
      bfrag[nn] =
          *(const v16bf*)(&Bs[cur][(nOff + nn * 16 + l16) * 32 + lh * 16]);

#pragma unroll
    for (int mm = 0; mm < 4; ++mm)
#pragma unroll
      for (int nn = 0; nn < 2; ++nn)
        acc[mm][nn] = wmma_bf16(afrag[mm], bfrag[nn], acc[mm][nn]);

#if HAVE_TDM
    if (more && wid == 0) __builtin_amdgcn_s_wait_tensorcnt(0);
#endif
    __syncthreads();
  }

  // --- epilogue: C/D layout VGPR r -> row r+8*lh, col l16 ---
  const size_t zoff = (size_t)z * (size_t)M * (size_t)N;
#pragma unroll
  for (int mm = 0; mm < 4; ++mm) {
#pragma unroll
    for (int nn = 0; nn < 2; ++nn) {
#pragma unroll
      for (int r = 0; r < 8; ++r) {
        int row = mBase + mOff + mm * 16 + r + 8 * lh;
        int col = nBase + nOff + nn * 16 + l16;
        float val = acc[mm][nn][r];
        if (BF16OUT) {
          ((__bf16*)Cout)[zoff + (size_t)row * N + col] = (__bf16)val;
        } else {
          ((float*)Cout)[zoff + (size_t)row * N + col] = val + bias[row];
        }
      }
    }
  }
}

// ---------------------------------------------------------------------------
// Attention stage, one block per (batch, head); 128 threads = 4 waves.
//   attn = softmax( (q . k^T) / 8 )   contracting over HW=4096 (WMMA bf16)
//   out[q,d] = rowsum(attn)[q] * sum_over_hd(V)[d]   (exact factorization of
//   the reference einsum 'bnqk,bnvd->bnqd')
// Fragments load straight from global: the contraction axis is contiguous.
// ---------------------------------------------------------------------------
__global__ __launch_bounds__(128)
void attn_kernel(const __bf16* __restrict__ qkv, __bf16* __restrict__ outbuf) {
  __shared__ float attn_s[64][65];
  __shared__ float S_s[64];
  __shared__ float vsum_s[HW];

  const int bn = blockIdx.x;
  const int b = bn >> 3, n = bn & 7;
  const __bf16* qb = qkv + (size_t)(b * M_QKV + n * HD) * HW;
  const __bf16* kb = qkv + (size_t)(b * M_QKV + CCH + n * HD) * HW;
  const __bf16* vb = qkv + (size_t)(b * M_QKV + 2 * CCH + n * HD) * HW;

  const int tid  = threadIdx.x;
  const int wid  = tid >> 5;
  const int lane = tid & 31;
  const int lh   = lane >> 4;
  const int l16  = lane & 15;
  const int mrow = wid * 16 + l16;

  v8f acc[4];
#pragma unroll
  for (int t = 0; t < 4; ++t) acc[t] = (v8f){0,0,0,0,0,0,0,0};

  const __bf16* ap = qb + (size_t)mrow * HW;
  for (int kk = 0; kk < HW; kk += 32) {
    v8bf lo = *(const v8bf*)(ap + kk + 16 * lh);
    v8bf hi = *(const v8bf*)(ap + kk + 16 * lh + 8);
    v16bf afrag = cat8(lo, hi);
#pragma unroll
    for (int t = 0; t < 4; ++t) {
      const __bf16* bp = kb + (size_t)(t * 16 + l16) * HW + kk + lh * 16;
      v16bf bfrag = *(const v16bf*)bp;
      acc[t] = wmma_bf16(afrag, bfrag, acc[t]);
    }
  }

  // dump scaled scores to LDS
#pragma unroll
  for (int t = 0; t < 4; ++t)
#pragma unroll
    for (int r = 0; r < 8; ++r)
      attn_s[wid * 16 + r + 8 * lh][t * 16 + l16] = acc[t][r] * 0.125f;
  __syncthreads();

  // softmax per row (threads 0..63), keep row sum of the softmax
  if (tid < 64) {
    float mx = -3.4e38f;
    for (int j = 0; j < 64; ++j) mx = fmaxf(mx, attn_s[tid][j]);
    float s = 0.f;
    for (int j = 0; j < 64; ++j) {
      float e = expf(attn_s[tid][j] - mx);
      attn_s[tid][j] = e;
      s += e;
    }
    float inv = 1.f / s, rs = 0.f;
    for (int j = 0; j < 64; ++j) { attn_s[tid][j] *= inv; rs += attn_s[tid][j]; }
    S_s[tid] = rs;
  }
  // column sums of V over the head dim (independent of softmax)
  for (int c = 0; c < 32; ++c) {
    int d = c * 128 + tid;
    float s = 0.f;
    for (int r = 0; r < 64; ++r) s += (float)vb[(size_t)r * HW + d];
    vsum_s[d] = s;
  }
  __syncthreads();

  // outer product write: out[q,d] = S[q] * vsum[d]
  __bf16* ob = outbuf + (size_t)(b * CCH + n * HD) * HW;
  for (int q = 0; q < 64; ++q) {
    float sq = S_s[q];
    for (int c = 0; c < 32; ++c) {
      int d = c * 128 + tid;
      ob[(size_t)q * HW + d] = (__bf16)(sq * vsum_s[d]);
    }
  }
}

// ---------------------------------------------------------------------------
// Launcher. Workspace layout (bytes):
//   X_bf16   : 16*512*4096*2   =  67,108,864
//   W1_bf16  : 1536*512*2      =   1,572,864
//   W2_bf16  : 512*512*2       =     524,288
//   QKV_bf16 : 16*1536*4096*2  = 201,326,592
//   OUT_bf16 : 16*512*4096*2   =  67,108,864   (total ~322 MB)
// ---------------------------------------------------------------------------
extern "C" void kernel_launch(void* const* d_in, const int* in_sizes, int n_in,
                              void* d_out, int out_size, void* d_ws,
                              size_t ws_size, hipStream_t stream) {
  const float* x      = (const float*)d_in[0];
  const float* w_qkv  = (const float*)d_in[1];
  const float* w_proj = (const float*)d_in[2];
  const float* b_proj = (const float*)d_in[3];
  float* y = (float*)d_out;

  char* ws = (char*)d_ws;
  const size_t X_OFF   = 0;
  const size_t W1_OFF  = X_OFF  + (size_t)BATCH * CCH * HW * 2;
  const size_t W2_OFF  = W1_OFF + (size_t)M_QKV * CCH * 2;
  const size_t QKV_OFF = W2_OFF + (size_t)CCH * CCH * 2;
  const size_t OUT_OFF = QKV_OFF + (size_t)BATCH * M_QKV * HW * 2;

  __bf16* Xb   = (__bf16*)(ws + X_OFF);
  __bf16* W1b  = (__bf16*)(ws + W1_OFF);
  __bf16* W2b  = (__bf16*)(ws + W2_OFF);
  __bf16* QKVb = (__bf16*)(ws + QKV_OFF);
  __bf16* OUTb = (__bf16*)(ws + OUT_OFF);

  // 0) fp32 -> bf16
  int nx = BATCH * CCH * HW;
  cvt_bf16_kernel<<<2048, 256, 0, stream>>>(x, Xb, nx);
  cvt_bf16_kernel<<<512, 256, 0, stream>>>(w_qkv, W1b, M_QKV * CCH);
  cvt_bf16_kernel<<<256, 256, 0, stream>>>(w_proj, W2b, CCH * CCH);

  // 1) qkv = W_qkv x X   (M=1536, N=4096, K=512, batched over 16)
  {
    dim3 grid(HW / 128, M_QKV / 128, BATCH);
    gemm_bf16_kernel<true><<<grid, 256, 0, stream>>>(
        W1b, Xb, (void*)QKVb, nullptr, M_QKV, HW, CCH);
  }

  // 2) attention per (batch, head)
  attn_kernel<<<BATCH * NH, 128, 0, stream>>>(QKVb, OUTb);

  // 3) y = W_proj x out + bias  (M=512, N=4096, K=512, batched over 16)
  {
    dim3 grid(HW / 128, CCH / 128, BATCH);
    gemm_bf16_kernel<false><<<grid, 256, 0, stream>>>(
        W2b, OUTb, (void*)y, b_proj, CCH, HW, CCH);
  }
}
